// PuzzleGNN_61718680043613
// MI455X (gfx1250) — compile-verified, matrix operand order checked by reading, not run on hardware
//
#include <hip/hip_runtime.h>
#include <hip/hip_bf16.h>
#include <math.h>

typedef __attribute__((ext_vector_type(16))) _Float16 v16h;
typedef __attribute__((ext_vector_type(8)))  float    v8f;

#define GH 4  // attention heads (fixed by reference)

// ---------------- helpers ----------------
static __device__ __forceinline__ unsigned fkey(float f) {
  unsigned u = __float_as_uint(f);
  return (u & 0x80000000u) ? ~u : (u | 0x80000000u);   // order-preserving map
}
static __device__ __forceinline__ float funkey(unsigned k) {
  unsigned u = (k & 0x80000000u) ? (k ^ 0x80000000u) : ~k;
  return __uint_as_float(u);
}
static __device__ __forceinline__ float lrelu(float v) { return v > 0.f ? v : 0.2f * v; }

// ---------------- small utility kernels ----------------
__global__ void k_fill_f32(float* __restrict__ p, int n, float v) {
  int i = blockIdx.x * blockDim.x + threadIdx.x;
  if (i < n) p[i] = v;
}
__global__ void k_fill_u32(unsigned* __restrict__ p, int n, unsigned v) {
  int i = blockIdx.x * blockDim.x + threadIdx.x;
  if (i < n) p[i] = v;
}
__global__ void k_build_edges(const int* __restrict__ ei, int E, int N,
                              int* __restrict__ src, int* __restrict__ dst) {
  int i = blockIdx.x * blockDim.x + threadIdx.x;
  if (i >= E + N) return;
  if (i < E) { src[i] = ei[i]; dst[i] = ei[E + i]; }
  else       { src[i] = i - E; dst[i] = i - E; }          // self loops
}
__global__ void k_deg(const int* __restrict__ dst, float* __restrict__ deg, int Etot) {
  int i = blockIdx.x * blockDim.x + threadIdx.x;
  if (i < Etot) atomicAdd(&deg[dst[i]], 1.0f);
}
__global__ void k_dis(float* __restrict__ deg, int N) {
  int i = blockIdx.x * blockDim.x + threadIdx.x;
  if (i >= N) return;
  float d = deg[i];
  deg[i] = d > 0.f ? rsqrtf(d) : 0.f;
}

// ---- WMMA GEMM: C[N,M] = A[N,K] * B[K,M] (f32 in, f16 MACs, f32 acc) ----
// Requires N%16==0 and M%16==0 (guaranteed by caller); K arbitrary.
__global__ __launch_bounds__(256)
void k_gemm_wmma(const float* __restrict__ A, const float* __restrict__ B,
                 float* __restrict__ C, int N, int K, int M) {
  __shared__ _Float16 sA[16 * 32];          // A tile, row-major [r][k]
  __shared__ _Float16 sB[8][16 * 32];       // per-wave B tile, col-major [n][k]
  const int t    = threadIdx.x;
  const int wid  = t >> 5;
  const int lane = t & 31;
  const int hi   = (lane >> 4) & 1;
  const int r    = lane & 15;
  const int row0 = blockIdx.x * 16;
  const int Mtiles = M >> 4;
  const int ct   = blockIdx.y * 8 + wid;
  const bool active = ct < Mtiles;
  const int col0 = ct * 16;
  // A staging: thread owns 2 consecutive k within one tile row
  const int arr = (2 * t) >> 5;             // tile row 0..15
  const int akk = (2 * t) & 31;             // even k within tile
  const float* arow = A + (size_t)(row0 + arr) * K;
  v8f acc = {};
  for (int k0 = 0; k0 < K; k0 += 32) {
    // ---- stage A: branchless clamp+select, no loops ----
    {
      int gk0 = k0 + akk, gk1 = gk0 + 1;
      float a0 = arow[gk0 < K ? gk0 : 0]; a0 = (gk0 < K) ? a0 : 0.f;
      float a1 = arow[gk1 < K ? gk1 : 0]; a1 = (gk1 < K) ? a1 : 0.f;
      sA[arr * 32 + akk]     = (_Float16)a0;
      sA[arr * 32 + akk + 1] = (_Float16)a1;
    }
    __syncthreads();
    if (active) {
      // ---- stage B: one k-row of 16 cols per lane, 4x float4 loads ----
      int gk = k0 + lane;
      int ck = (gk < K) ? gk : 0;
      float sc = (gk < K) ? 1.f : 0.f;      // zero-pad K tail via multiply
      const float4* wp = (const float4*)(B + (size_t)ck * M + col0);
      float4 w0 = wp[0], w1 = wp[1], w2 = wp[2], w3 = wp[3];
      float wv[16] = { w0.x, w0.y, w0.z, w0.w,  w1.x, w1.y, w1.z, w1.w,
                       w2.x, w2.y, w2.z, w2.w,  w3.x, w3.y, w3.z, w3.w };
#pragma unroll
      for (int n = 0; n < 16; ++n)
        sB[wid][n * 32 + lane] = (_Float16)(wv[n] * sc);
      // ---- fragments per ISA 16-bit packing:
      // lane holds row/col (lane&15); elems 0..7 -> K=hi*8+0..7, elems 8..15 -> K=16+hi*8+0..7
      v16h av, bv;
      const _Float16* ar = &sA[r * 32];
      const _Float16* br = &sB[wid][r * 32];
#pragma unroll
      for (int i = 0; i < 8; ++i) {
        av[i]     = ar[hi * 8 + i];
        av[8 + i] = ar[16 + hi * 8 + i];
        bv[i]     = br[hi * 8 + i];
        bv[8 + i] = br[16 + hi * 8 + i];
      }
      acc = __builtin_amdgcn_wmma_f32_16x16x32_f16(false, av, false, bv,
                                                   (short)0, acc, false, false);
    }
    __syncthreads();
  }
  if (active) {
    int gc = col0 + r;
#pragma unroll
    for (int i = 0; i < 8; ++i)                 // C layout: VGPR i -> row hi*8+i
      C[(size_t)(row0 + hi * 8 + i) * M + gc] = acc[i];
  }
}

// ---------------- GAT kernels ----------------
__global__ void k_gat_scores(const float* __restrict__ h, const float* __restrict__ as,
                             const float* __restrict__ ad, float* __restrict__ es,
                             float* __restrict__ ed, int N, int F) {
  int i = blockIdx.x * blockDim.x + threadIdx.x;
  if (i >= N * GH) return;
  int n = i >> 2, hh = i & 3;
  const float* hp  = h + (size_t)n * (GH * F) + hh * F;
  const float* asp = as + hh * F;
  const float* adp = ad + hh * F;
  float s0 = 0.f, s1 = 0.f;
  for (int f = 0; f < F; ++f) { float hv = hp[f]; s0 += hv * asp[f]; s1 += hv * adp[f]; }
  es[i] = s0; ed[i] = s1;
}
__global__ void k_gat_emax(const int* __restrict__ src, const int* __restrict__ dst,
                           const float* __restrict__ es, const float* __restrict__ ed,
                           unsigned* __restrict__ mkey, int Etot) {
  int i = blockIdx.x * blockDim.x + threadIdx.x;
  if (i >= Etot * GH) return;
  int e = i >> 2, hh = i & 3;
  float v = lrelu(es[src[e] * GH + hh] + ed[dst[e] * GH + hh]);
  atomicMax(&mkey[dst[e] * GH + hh], fkey(v));
}
__global__ void k_gat_esum(const int* __restrict__ src, const int* __restrict__ dst,
                           const float* __restrict__ es, const float* __restrict__ ed,
                           const unsigned* __restrict__ mkey, float* __restrict__ ssum,
                           int Etot) {
  int i = blockIdx.x * blockDim.x + threadIdx.x;
  if (i >= Etot * GH) return;
  int e = i >> 2, hh = i & 3;
  int di = dst[e] * GH + hh;
  float v = lrelu(es[src[e] * GH + hh] + ed[di]);
  atomicAdd(&ssum[di], expf(v - funkey(mkey[di])));
}
__global__ void k_gat_alpha(const int* __restrict__ src, const int* __restrict__ dst,
                            const float* __restrict__ es, const float* __restrict__ ed,
                            const unsigned* __restrict__ mkey, const float* __restrict__ ssum,
                            float* __restrict__ alpha, int Etot) {
  int i = blockIdx.x * blockDim.x + threadIdx.x;
  if (i >= Etot * GH) return;
  int e = i >> 2, hh = i & 3;
  int di = dst[e] * GH + hh;
  float v = lrelu(es[src[e] * GH + hh] + ed[di]);
  alpha[i] = expf(v - funkey(mkey[di])) / (ssum[di] + 1e-16f);
}
// one thread handles 4 consecutive features of one edge (float4 load + 4 atomics)
__global__ void k_gat_scatter4(const int* __restrict__ src, const int* __restrict__ dst,
                               const float* __restrict__ h, const float* __restrict__ alpha,
                               float* __restrict__ acc, int Etot, int mshift, int fshift) {
  int i = blockIdx.x * blockDim.x + threadIdx.x;
  int qshift = mshift - 2;
  int total = Etot << qshift;
  if (i >= total) return;
  int e = i >> qshift;
  int m = (i & ((1 << qshift) - 1)) << 2;
  int hh = m >> fshift;                       // F>=16 so all 4 share one head
  float a = alpha[e * GH + hh];
  const float4 hv = *(const float4*)(h + ((size_t)src[e] << mshift) + m);
  float* ap = acc + ((size_t)dst[e] << mshift) + m;
  atomicAdd(ap + 0, hv.x * a);
  atomicAdd(ap + 1, hv.y * a);
  atomicAdd(ap + 2, hv.z * a);
  atomicAdd(ap + 3, hv.w * a);
}
__global__ void k_gat_final(const float* __restrict__ acc, const float* __restrict__ b,
                            float* __restrict__ xout, int N, int fshift) {
  int i = blockIdx.x * blockDim.x + threadIdx.x;
  int F = 1 << fshift;
  if (i >= N * F) return;
  int n = i >> fshift, f = i & (F - 1);
  const float* ap = acc + ((size_t)n << (fshift + 2));   // M = 4F
  float v = 0.25f * (ap[f] + ap[F + f] + ap[2 * F + f] + ap[3 * F + f]) + b[f];
  xout[i] = v > 0.f ? v : 0.f;
}

// ---------------- GCN kernels ----------------
__global__ void k_gcn_scatter4(const int* __restrict__ src, const int* __restrict__ dst,
                               const float* __restrict__ h, const float* __restrict__ dis,
                               float* __restrict__ acc, int Etot, int mshift) {
  int i = blockIdx.x * blockDim.x + threadIdx.x;
  int qshift = mshift - 2;
  int total = Etot << qshift;
  if (i >= total) return;
  int e = i >> qshift;
  int m = (i & ((1 << qshift) - 1)) << 2;
  int s = src[e], d = dst[e];
  float w = dis[s] * dis[d];
  const float4 hv = *(const float4*)(h + ((size_t)s << mshift) + m);
  float* ap = acc + ((size_t)d << mshift) + m;
  atomicAdd(ap + 0, hv.x * w);
  atomicAdd(ap + 1, hv.y * w);
  atomicAdd(ap + 2, hv.z * w);
  atomicAdd(ap + 3, hv.w * w);
}
__global__ void k_gcn_final(const float* __restrict__ acc, const float* __restrict__ b,
                            float* __restrict__ xout, int N, int mshift) {
  int i = blockIdx.x * blockDim.x + threadIdx.x;
  int M = 1 << mshift;
  if (i >= N * M) return;
  int m = i & (M - 1);
  float v = acc[i] + b[m];
  xout[i] = v > 0.f ? v : 0.f;
}

// ---------------- fused MLP head ----------------
__global__ void k_mlp(const float* __restrict__ x,
                      const float* __restrict__ W1, const float* __restrict__ b1,
                      const float* __restrict__ W2, const float* __restrict__ b2,
                      const float* __restrict__ W3, const float* __restrict__ b3,
                      const float* __restrict__ W4, const float* __restrict__ b4,
                      float* __restrict__ out, int N) {
  int n = blockIdx.x * blockDim.x + threadIdx.x;
  if (n >= N) return;
  float h0[16], h1[8], h2[8], h3[4];
#pragma unroll
  for (int i = 0; i < 16; ++i) h0[i] = x[(size_t)n * 16 + i];
#pragma unroll
  for (int j = 0; j < 8; ++j) { float s = b1[j];
#pragma unroll
    for (int i = 0; i < 16; ++i) s += h0[i] * W1[i * 8 + j];
    h1[j] = fmaxf(s, 0.f); }
#pragma unroll
  for (int j = 0; j < 8; ++j) { float s = b2[j];
#pragma unroll
    for (int i = 0; i < 8; ++i) s += h1[i] * W2[i * 8 + j];
    h2[j] = fmaxf(s, 0.f); }
#pragma unroll
  for (int j = 0; j < 4; ++j) { float s = b3[j];
#pragma unroll
    for (int i = 0; i < 8; ++i) s += h2[i] * W3[i * 4 + j];
    h3[j] = fmaxf(s, 0.f); }
  float s = b4[0];
#pragma unroll
  for (int i = 0; i < 4; ++i) s += h3[i] * W4[i];
  out[n] = rintf(s);                       // jnp.round = nearest-even
}

// ---------------- host orchestration ----------------
extern "C" void kernel_launch(void* const* d_in, const int* in_sizes, int n_in,
                              void* d_out, int out_size, void* d_ws, size_t ws_size,
                              hipStream_t stream) {
  (void)n_in; (void)out_size; (void)ws_size;
  const int N    = in_sizes[0] / 5;
  const int E    = in_sizes[1] / 2;
  const int Etot = E + N;

  const float* x0 = (const float*)d_in[0];
  const int*   ei = (const int*)d_in[1];
  const float* g1_W = (const float*)d_in[2];  const float* g1_as = (const float*)d_in[3];
  const float* g1_ad = (const float*)d_in[4]; const float* g1_b  = (const float*)d_in[5];
  const float* g2_W = (const float*)d_in[6];  const float* g2_as = (const float*)d_in[7];
  const float* g2_ad = (const float*)d_in[8]; const float* g2_b  = (const float*)d_in[9];
  const float* c3_W = (const float*)d_in[10]; const float* c3_b  = (const float*)d_in[11];
  const float* g4_W = (const float*)d_in[12]; const float* g4_as = (const float*)d_in[13];
  const float* g4_ad = (const float*)d_in[14];const float* g4_b  = (const float*)d_in[15];
  const float* g5_W = (const float*)d_in[16]; const float* g5_as = (const float*)d_in[17];
  const float* g5_ad = (const float*)d_in[18];const float* g5_b  = (const float*)d_in[19];
  const float* c6_W = (const float*)d_in[20]; const float* c6_b  = (const float*)d_in[21];
  const float* g7_W = (const float*)d_in[22]; const float* g7_as = (const float*)d_in[23];
  const float* g7_ad = (const float*)d_in[24];const float* g7_b  = (const float*)d_in[25];
  const float* c8_W = (const float*)d_in[26]; const float* c8_b  = (const float*)d_in[27];
  const float* c9_W = (const float*)d_in[28]; const float* c9_b  = (const float*)d_in[29];
  const float* m1_W = (const float*)d_in[30]; const float* m1_b  = (const float*)d_in[31];
  const float* m2_W = (const float*)d_in[32]; const float* m2_b  = (const float*)d_in[33];
  const float* m3_W = (const float*)d_in[34]; const float* m3_b  = (const float*)d_in[35];
  const float* m4_W = (const float*)d_in[36]; const float* m4_b  = (const float*)d_in[37];

  // workspace bump allocator
  size_t off = 0;
  auto alloc = [&](size_t bytes) -> void* {
    bytes = (bytes + 255) & ~(size_t)255;
    void* p = (char*)d_ws + off; off += bytes; return p;
  };
  int*      srcv  = (int*)alloc((size_t)Etot * 4);
  int*      dstv  = (int*)alloc((size_t)Etot * 4);
  float*    disb  = (float*)alloc((size_t)N * 4);
  float*    hbuf  = (float*)alloc((size_t)N * 512 * 4);
  float*    accb  = (float*)alloc((size_t)N * 512 * 4);
  float*    xA    = (float*)alloc((size_t)N * 128 * 4);
  float*    xB    = (float*)alloc((size_t)N * 128 * 4);
  float*    esb   = (float*)alloc((size_t)N * GH * 4);
  float*    edb   = (float*)alloc((size_t)N * GH * 4);
  unsigned* mkey  = (unsigned*)alloc((size_t)N * GH * 4);
  float*    ssum  = (float*)alloc((size_t)N * GH * 4);
  float*    alphab= (float*)alloc((size_t)Etot * GH * 4);

  auto cdiv  = [](int a, int b) { return (a + b - 1) / b; };
  auto ilog2 = [](int v) { int s = 0; while ((1 << s) < v) ++s; return s; };
  auto fillf = [&](float* p, int n, float v) { k_fill_f32<<<cdiv(n, 256), 256, 0, stream>>>(p, n, v); };
  auto fillu = [&](unsigned* p, int n, unsigned v) { k_fill_u32<<<cdiv(n, 256), 256, 0, stream>>>(p, n, v); };
  auto gemm  = [&](const float* A, const float* W, float* C, int K, int M) {
    dim3 grid(cdiv(N, 16), cdiv(cdiv(M, 16), 8));
    k_gemm_wmma<<<grid, 256, 0, stream>>>(A, W, C, N, K, M);
  };

  // graph prep (edges + symmetric-norm degrees, reused by every layer)
  k_build_edges<<<cdiv(Etot, 256), 256, 0, stream>>>(ei, E, N, srcv, dstv);
  fillf(disb, N, 0.f);
  k_deg<<<cdiv(Etot, 256), 256, 0, stream>>>(dstv, disb, Etot);
  k_dis<<<cdiv(N, 256), 256, 0, stream>>>(disb, N);

  auto run_gat = [&](const float* xin, int K, const float* W, const float* as,
                     const float* ad, const float* b, int F, float* xout) {
    int M = GH * F, ms = ilog2(M), fs = ilog2(F);
    gemm(xin, W, hbuf, K, M);
    k_gat_scores<<<cdiv(N * GH, 256), 256, 0, stream>>>(hbuf, as, ad, esb, edb, N, F);
    fillu(mkey, N * GH, 0x007FFFFFu);         // fkey(-inf)
    fillf(ssum, N * GH, 0.f);
    fillf(accb, N * M, 0.f);
    k_gat_emax <<<cdiv(Etot * GH, 256), 256, 0, stream>>>(srcv, dstv, esb, edb, mkey, Etot);
    k_gat_esum <<<cdiv(Etot * GH, 256), 256, 0, stream>>>(srcv, dstv, esb, edb, mkey, ssum, Etot);
    k_gat_alpha<<<cdiv(Etot * GH, 256), 256, 0, stream>>>(srcv, dstv, esb, edb, mkey, ssum, alphab, Etot);
    k_gat_scatter4<<<cdiv(Etot * (M >> 2), 256), 256, 0, stream>>>(srcv, dstv, hbuf, alphab, accb,
                                                                   Etot, ms, fs);
    k_gat_final<<<cdiv(N * F, 256), 256, 0, stream>>>(accb, b, xout, N, fs);
  };
  auto run_gcn = [&](const float* xin, int K, const float* W, const float* b, int M, float* xout) {
    int ms = ilog2(M);
    gemm(xin, W, hbuf, K, M);
    fillf(accb, N * M, 0.f);
    k_gcn_scatter4<<<cdiv(Etot * (M >> 2), 256), 256, 0, stream>>>(srcv, dstv, hbuf, disb, accb,
                                                                   Etot, ms);
    k_gcn_final<<<cdiv(N * M, 256), 256, 0, stream>>>(accb, b, xout, N, ms);
  };

  run_gat(x0, 5,   g1_W, g1_as, g1_ad, g1_b, 16,  xA);   // -> [N,16]
  run_gat(xA, 16,  g2_W, g2_as, g2_ad, g2_b, 64,  xB);   // -> [N,64]
  run_gcn(xB, 64,  c3_W, c3_b, 64,  xA);                 // -> [N,64]
  run_gat(xA, 64,  g4_W, g4_as, g4_ad, g4_b, 128, xB);   // -> [N,128]
  run_gat(xB, 128, g5_W, g5_as, g5_ad, g5_b, 64,  xA);   // -> [N,64]
  run_gcn(xA, 64,  c6_W, c6_b, 64,  xB);                 // -> [N,64]
  run_gat(xB, 64,  g7_W, g7_as, g7_ad, g7_b, 32,  xA);   // -> [N,32]
  run_gcn(xA, 32,  c8_W, c8_b, 32,  xB);                 // -> [N,32]
  run_gcn(xB, 32,  c9_W, c9_b, 16,  xA);                 // -> [N,16]

  k_mlp<<<cdiv(N, 256), 256, 0, stream>>>(xA, m1_W, m1_b, m2_W, m2_b,
                                          m3_W, m3_b, m4_W, m4_b,
                                          (float*)d_out, N);
}
